// MOERouter_26448408609192
// MI455X (gfx1250) — matrix operand order repaired and checked by reading, not compiled, not run on hardware
//
#include <hip/hip_runtime.h>
#include <hip/hip_bf16.h>
#include <stdint.h>

// ---------------------------------------------------------------------------
// MoE router, fused: logits = (x @ W1^T + b1) @ W2^T + b2, softmax, top-2,
// renormalized weights, one-hot expert masks.
//
// GEMM1: v_wmma_f32_16x16x32_bf16 (fp32 accumulate).
//   - W1 is pre-converted to bf16 in d_ws (1 MB) so B fragments are raw
//     32-byte bf16 loads (no per-wave convert, WGP$-resident K-slab).
//   - x k-slabs are staged fp32 -> LDS with global_load_async_to_lds_b128
//     (ASYNCcnt-tracked, double buffered, drained with s_wait_asynccnt).
// GEMM2 (E=8) folded in registers + lane-shuffle reduction.
// ---------------------------------------------------------------------------

typedef __attribute__((ext_vector_type(16))) __bf16 v16bf;
typedef __attribute__((ext_vector_type(8)))  __bf16 v8bf;
typedef __attribute__((ext_vector_type(8)))  float  v8f;

#define N_TOK      32768
#define DIM        1024
#define HDIM       512
#define NEXP       8
#define TOK_PER_WG 64
#define XPAD       36      // padded K-slab row (floats): 144B, 16B aligned

// -------- CDNA5 async copy: 16 bytes global -> LDS, per lane ---------------
__device__ __forceinline__ void async_cp16(unsigned lds_off, const float* g) {
    asm volatile("global_load_async_to_lds_b128 %0, %1, off"
                 :: "v"(lds_off), "v"(g) : "memory");
}
__device__ __forceinline__ void async_drain() {
    asm volatile("s_wait_asynccnt 0x0" ::: "memory");
}

// -------- W1 fp32 -> bf16 pre-convert (into d_ws) --------------------------
__global__ __launch_bounds__(256)
void convert_w1_kernel(const float* __restrict__ W1,
                       unsigned short* __restrict__ W1bf)
{
    const size_t i = ((size_t)blockIdx.x * 256 + threadIdx.x) * 8;
    const float4 f0 = *(const float4*)(W1 + i);
    const float4 f1 = *(const float4*)(W1 + i + 4);
    v8bf o;
    o[0] = (__bf16)f0.x; o[1] = (__bf16)f0.y;
    o[2] = (__bf16)f0.z; o[3] = (__bf16)f0.w;
    o[4] = (__bf16)f1.x; o[5] = (__bf16)f1.y;
    o[6] = (__bf16)f1.z; o[7] = (__bf16)f1.w;
    *reinterpret_cast<v8bf*>(W1bf + i) = o;
}

// -------- fused router kernel ----------------------------------------------
template<bool BPRE>
__global__ __launch_bounds__(256)
void moe_router_kernel(const float* __restrict__ x,
                       const float* __restrict__ W1,
                       const float* __restrict__ b1,
                       const float* __restrict__ W2,
                       const float* __restrict__ b2,
                       const unsigned short* __restrict__ W1bf,
                       float* __restrict__ out)
{
    // double-buffered fp32 x k-slab (filled by async DMA) + logit partials
    __shared__ float xs[2][TOK_PER_WG][XPAD];
    __shared__ float lbuf[2][TOK_PER_WG][NEXP];

    const int tid   = threadIdx.x;
    const int lane  = tid & 31;
    const int wave  = tid >> 5;
    const int tsub  = wave & 3;    // token subtile (16 tokens each)
    const int fhalf = wave >> 2;   // feature half (256 features each)

    const int l15 = lane & 15;
    const int hi  = lane >> 4;     // 0 (lanes 0-15) or 1 (lanes 16-31)

    const int blockBase = blockIdx.x * TOK_PER_WG;
    const int fBase     = fhalf * 256;

    // staging assignment: thread copies 8 floats of one token row
    const int stok = tid >> 2;           // 0..63
    const int skq  = (tid & 3) << 3;     // 0,8,16,24

    // ---------------- GEMM1: h = x @ W1^T (bf16 WMMA, f32 accum) -----------
    v8f acc[16];
#pragma unroll
    for (int j = 0; j < 16; ++j)
        acc[j] = (v8f){0.f, 0.f, 0.f, 0.f, 0.f, 0.f, 0.f, 0.f};

    // prologue: stage k-slab 0 into buffer 0
    {
        const float* g = x + (size_t)(blockBase + stok) * DIM + skq;
        const unsigned off0 = (unsigned)(uintptr_t)(&xs[0][stok][skq]);
        async_cp16(off0, g);
        async_cp16(off0 + 16, g + 4);
    }
    async_drain();
    __syncthreads();

    int cur = 0;
    for (int kk = 0; kk < DIM; kk += 32) {
        // issue async copy of the next slab into the other buffer
        if (kk + 32 < DIM) {
            const float* g =
                x + (size_t)(blockBase + stok) * DIM + (kk + 32) + skq;
            const unsigned off0 =
                (unsigned)(uintptr_t)(&xs[cur ^ 1][stok][skq]);
            async_cp16(off0, g);
            async_cp16(off0 + 16, g + 4);
        }

        // ---- A fragment from LDS slab (contiguous k order, same as B) ----
        // low lanes: K = 0..15; high lanes: K = 16..31
        const float* arow = &xs[cur][tsub * 16 + l15][0];
        const int ka = hi << 4;
        const float4 a0 = *(const float4*)(arow + ka);
        const float4 a1 = *(const float4*)(arow + ka + 4);
        const float4 a2 = *(const float4*)(arow + ka + 8);
        const float4 a3 = *(const float4*)(arow + ka + 12);

        v16bf a;
        a[0]  = (__bf16)a0.x; a[1]  = (__bf16)a0.y;
        a[2]  = (__bf16)a0.z; a[3]  = (__bf16)a0.w;
        a[4]  = (__bf16)a1.x; a[5]  = (__bf16)a1.y;
        a[6]  = (__bf16)a1.z; a[7]  = (__bf16)a1.w;
        a[8]  = (__bf16)a2.x; a[9]  = (__bf16)a2.y;
        a[10] = (__bf16)a2.z; a[11] = (__bf16)a2.w;
        a[12] = (__bf16)a3.x; a[13] = (__bf16)a3.y;
        a[14] = (__bf16)a3.z; a[15] = (__bf16)a3.w;

        // ---- 16 N-tiles of B = W1^T (32x16, contiguous k order as in A) ----
        // lane = column n; low lanes K = kk+0..15, high lanes K = kk+16..31
        const int kB = kk + (hi << 4);
#pragma unroll
        for (int j = 0; j < 16; ++j) {
            const int frow = fBase + j * 16 + l15;
            v16bf b;
            if constexpr (BPRE) {
                // raw bf16 fragment: one 32-byte load, no convert
                const unsigned short* wrow = W1bf + (size_t)frow * DIM + kB;
                __builtin_prefetch(wrow + 32, 0, 1);   // global_prefetch_b8
                b = *reinterpret_cast<const v16bf*>(wrow);
            } else {
                const float* wrow = W1 + (size_t)frow * DIM + kB;
                const float4 c0 = *(const float4*)(wrow);
                const float4 c1 = *(const float4*)(wrow + 4);
                const float4 c2 = *(const float4*)(wrow + 8);
                const float4 c3 = *(const float4*)(wrow + 12);
                b[0]  = (__bf16)c0.x; b[1]  = (__bf16)c0.y;
                b[2]  = (__bf16)c0.z; b[3]  = (__bf16)c0.w;
                b[4]  = (__bf16)c1.x; b[5]  = (__bf16)c1.y;
                b[6]  = (__bf16)c1.z; b[7]  = (__bf16)c1.w;
                b[8]  = (__bf16)c2.x; b[9]  = (__bf16)c2.y;
                b[10] = (__bf16)c2.z; b[11] = (__bf16)c2.w;
                b[12] = (__bf16)c3.x; b[13] = (__bf16)c3.y;
                b[14] = (__bf16)c3.z; b[15] = (__bf16)c3.w;
            }
            acc[j] = __builtin_amdgcn_wmma_f32_16x16x32_bf16(
                false, a, false, b, (short)0, acc[j], false, false);
        }

        // make the freshly staged slab visible to the whole workgroup
        async_drain();
        __syncthreads();
        cur ^= 1;
    }

    // ------------- GEMM2 fold: logit partials from registers ---------------
    // C layout: VGPR r, lanes 0-15 -> token r, lanes 16-31 -> token 8+r;
    // feature = fBase + j*16 + (lane & 15).
    float plog[8][NEXP];
#pragma unroll
    for (int r = 0; r < 8; ++r)
#pragma unroll
        for (int e = 0; e < NEXP; ++e)
            plog[r][e] = 0.f;

#pragma unroll
    for (int j = 0; j < 16; ++j) {
        const int   f   = fBase + j * 16 + l15;
        const float b1v = b1[f];
        float hv[8];
#pragma unroll
        for (int r = 0; r < 8; ++r)
            hv[r] = acc[j][r] + b1v;
#pragma unroll
        for (int e = 0; e < NEXP; ++e) {
            const float w2v = W2[(size_t)e * HDIM + f];
#pragma unroll
            for (int r = 0; r < 8; ++r)
                plog[r][e] = fmaf(hv[r], w2v, plog[r][e]);
        }
    }

    // reduce across the 16 feature lanes inside each half-wave
#pragma unroll
    for (int off = 8; off >= 1; off >>= 1) {
#pragma unroll
        for (int r = 0; r < 8; ++r)
#pragma unroll
            for (int e = 0; e < NEXP; ++e)
                plog[r][e] += __shfl_xor(plog[r][e], off, 16);
    }

    if (l15 == 0) {
        const int t0 = tsub * 16 + hi * 8;
#pragma unroll
        for (int r = 0; r < 8; ++r)
#pragma unroll
            for (int e = 0; e < NEXP; ++e)
                lbuf[fhalf][t0 + r][e] = plog[r][e];
    }
    __syncthreads();

    // ---------------- Phase C: softmax / top-2 / outputs -------------------
    if (tid < TOK_PER_WG) {
        const int tok = blockBase + tid;

        float l[NEXP];
#pragma unroll
        for (int e = 0; e < NEXP; ++e)
            l[e] = lbuf[0][tid][e] + lbuf[1][tid][e] + b2[e];

        float* out_logits = out;
        float* out_w      = out + (size_t)N_TOK * NEXP;
        int*   out_idx    = (int*)(out + (size_t)N_TOK * (NEXP + 2));
        int*   out_masks  = (int*)(out + (size_t)N_TOK * (NEXP + 4));

#pragma unroll
        for (int e = 0; e < NEXP; ++e)
            out_logits[(size_t)tok * NEXP + e] = l[e];

        // top-1 (lowest index wins ties, matching jax.lax.top_k)
        int i0 = 0; float v0 = l[0];
#pragma unroll
        for (int e = 1; e < NEXP; ++e)
            if (l[e] > v0) { v0 = l[e]; i0 = e; }
        // top-2
        int   i1 = (i0 == 0) ? 1 : 0;
        float v1 = (i0 == 0) ? l[1] : l[0];
#pragma unroll
        for (int e = 0; e < NEXP; ++e)
            if (e != i0 && l[e] > v1) { v1 = l[e]; i1 = e; }

        // renormalized top-2 weights == stable 2-way softmax of (v0, v1)
        const float t   = __expf(v1 - v0);
        const float inv = 1.0f / (1.0f + t);
        out_w[(size_t)tok * 2 + 0] = inv;
        out_w[(size_t)tok * 2 + 1] = t * inv;
        out_idx[(size_t)tok * 2 + 0] = i0;
        out_idx[(size_t)tok * 2 + 1] = i1;

        // expert_masks [E, K, N]
#pragma unroll
        for (int e = 0; e < NEXP; ++e) {
            out_masks[(size_t)(e * 2 + 0) * N_TOK + tok] = (i0 == e) ? 1 : 0;
            out_masks[(size_t)(e * 2 + 1) * N_TOK + tok] = (i1 == e) ? 1 : 0;
        }
    }
}

extern "C" void kernel_launch(void* const* d_in, const int* in_sizes, int n_in,
                              void* d_out, int out_size, void* d_ws, size_t ws_size,
                              hipStream_t stream) {
    (void)in_sizes; (void)n_in; (void)out_size;
    const float* x  = (const float*)d_in[0];
    const float* W1 = (const float*)d_in[1];
    const float* b1 = (const float*)d_in[2];
    const float* W2 = (const float*)d_in[3];
    const float* b2 = (const float*)d_in[4];

    dim3 grid(N_TOK / TOK_PER_WG);   // 512 workgroups
    dim3 block(256);                 // 8 wave32s

    const size_t w1bf_bytes = (size_t)HDIM * DIM * sizeof(unsigned short);
    if (ws_size >= w1bf_bytes) {
        unsigned short* W1bf = (unsigned short*)d_ws;
        // 512*1024 elements / (256 threads * 8 per thread) = 256 blocks
        convert_w1_kernel<<<dim3(256), dim3(256), 0, stream>>>(W1, W1bf);
        moe_router_kernel<true><<<grid, block, 0, stream>>>(
            x, W1, b1, W2, b2, W1bf, (float*)d_out);
    } else {
        moe_router_kernel<false><<<grid, block, 0, stream>>>(
            x, W1, b1, W2, b2, nullptr, (float*)d_out);
    }
}
